// ErnieLayoutSelfAttention_12163347382760
// MI455X (gfx1250) — matrix operand order repaired and checked by reading, not compiled, not run on hardware
//
#include <hip/hip_runtime.h>

// Problem constants
#define BB  4
#define SS  1024
#define HH  768
#define NHH 12
#define HDD 64

typedef _Float16 h8   __attribute__((ext_vector_type(8)));
typedef _Float16 v16h __attribute__((ext_vector_type(16)));
typedef float    v8f  __attribute__((ext_vector_type(8)));

// Assemble A-operand (16x32 f16) fragment from two contiguous 8-half chunks.
// Layout: lane L (g=L/16, m=L%16) holds K = [g*16, g*16+8) in VGPR0-3 and
// K = [g*16+8, g*16+16) in VGPR4-7 (contiguous, matches B frags).
__device__ __forceinline__ v16h make_a(h8 lo, h8 hi) {
    v16h r;
#pragma unroll
    for (int i = 0; i < 8; i++) { r[i] = lo[i]; r[i + 8] = hi[i]; }
    return r;
}

__device__ __forceinline__ v16h load_a_frag(const _Float16* row, int g) {
    h8 lo = *(const h8*)(row + g * 16);
    h8 hi = *(const h8*)(row + g * 16 + 8);
    return make_a(lo, hi);
}

__device__ __forceinline__ v8f wmma16(v16h a, v16h b, v8f c) {
    return __builtin_amdgcn_wmma_f32_16x16x32_f16(false, a, false, b,
                                                  (short)0, c, false, false);
}

// ---------------- Kernel 1: hidden_states fp32 -> f16 ----------------
__global__ __launch_bounds__(256) void prep_hs(const float* __restrict__ hs,
                                               _Float16* __restrict__ hs16) {
    int i = blockIdx.x * 256 + threadIdx.x;   // exactly 4096*768 threads
    hs16[i] = (_Float16)hs[i];
}

// ------- Kernel 2: Wq/Wk/Wv fp32 [K][N] -> f16 transposed Wt[w][N][K] -------
__global__ __launch_bounds__(256) void prep_w(const float* __restrict__ Wq,
                                              const float* __restrict__ Wk,
                                              const float* __restrict__ Wv,
                                              _Float16* __restrict__ Wt) {
    int i = blockIdx.x * 256 + threadIdx.x;   // exactly 3*768*768 threads
    int w = i / (HH * HH);
    int r = i % (HH * HH);
    int n = r / HH;
    int k = r % HH;
    const float* W = (w == 0) ? Wq : ((w == 1) ? Wk : Wv);
    Wt[i] = (_Float16)W[k * HH + n];
}

// ---------------- Kernel 3: QKV projection GEMM (WMMA f16) ----------------
// One wave -> 16(M) x 64(N) tile; N chunk of 64 == one head.
// Software-pipelined: k-slab ks+1 frag loads issue before slab ks's WMMAs.
__global__ __launch_bounds__(256) void qkv_gemm(
    const _Float16* __restrict__ hs16, const _Float16* __restrict__ Wt16,
    const float* __restrict__ bq, const float* __restrict__ bk,
    const float* __restrict__ bv,
    _Float16* __restrict__ q16, _Float16* __restrict__ k16,
    _Float16* __restrict__ vt16) {
    const int wid  = blockIdx.x * 8 + (threadIdx.x >> 5);  // 9216 wave tasks
    const int lane = threadIdx.x & 31;
    const int mat  = wid / (256 * NHH);        // 0=Q 1=K 2=V
    const int rem  = wid % (256 * NHH);
    const int m0   = (rem / NHH) * 16;         // global row tile in [0,4096)
    const int head = rem % NHH;
    const int nbase = head * HDD;
    const int g  = lane >> 4;
    const int nl = lane & 15;

    const _Float16* A = hs16 + (size_t)(m0 + nl) * HH;  // lane's A row (m = nl)
    const _Float16* Bt[4];
#pragma unroll
    for (int ht = 0; ht < 4; ht++)
        Bt[ht] = Wt16 + (size_t)mat * HH * HH + (size_t)(nbase + ht * 16 + nl) * HH;

    v8f c[4] = {};

    // prologue: slab 0 frags
    v16h ac = load_a_frag(A, g);
    v16h bc[4];
#pragma unroll
    for (int ht = 0; ht < 4; ht++) bc[ht] = *(const v16h*)(Bt[ht] + g * 16);

#pragma unroll 2
    for (int ks = 0; ks < HH / 32; ks++) {
        v16h an = {};
        v16h bn[4] = {};
        if (ks < HH / 32 - 1) {                 // issue next-slab loads first
            const int kn = (ks + 1) * 32;
            an = load_a_frag(A + kn, g);
#pragma unroll
            for (int ht = 0; ht < 4; ht++)
                bn[ht] = *(const v16h*)(Bt[ht] + kn + g * 16);
        }
#pragma unroll
        for (int ht = 0; ht < 4; ht++)          // consume resident frags
            c[ht] = wmma16(ac, bc[ht], c[ht]);
        ac = an;
#pragma unroll
        for (int ht = 0; ht < 4; ht++) bc[ht] = bn[ht];
    }

    const float* bias = (mat == 0) ? bq : ((mat == 1) ? bk : bv);
    const int b  = m0 >> 10;        // S = 1024
    const int s0 = m0 & (SS - 1);

    if (mat < 2) {
        _Float16* dst = (mat == 0 ? q16 : k16) + (size_t)(b * NHH + head) * SS * HDD;
#pragma unroll
        for (int ht = 0; ht < 4; ht++) {
            float bb = bias[nbase + ht * 16 + nl];
#pragma unroll
            for (int r = 0; r < 8; r++) {
                int s = s0 + 8 * g + r;   // C layout: row m = r + 8g
                dst[(size_t)s * HDD + ht * 16 + nl] = (_Float16)(c[ht][r] + bb);
            }
        }
    } else {
        _Float16* dst = vt16 + (size_t)(b * NHH + head) * HDD * SS;
#pragma unroll
        for (int ht = 0; ht < 4; ht++) {
            float bb = bias[nbase + ht * 16 + nl];
            h8 pack;
#pragma unroll
            for (int r = 0; r < 8; r++) pack[r] = (_Float16)(c[ht][r] + bb);
            // rows r are contiguous in S -> single 16B store
            *(h8*)(dst + (size_t)(ht * 16 + nl) * SS + s0 + 8 * g) = pack;
        }
    }
}

// ---------------- Kernel 4: fused flash attention ----------------
__global__ __launch_bounds__(256) void attn(
    const _Float16* __restrict__ q16, const _Float16* __restrict__ k16,
    const _Float16* __restrict__ vt16,
    const float* __restrict__ rel, const float* __restrict__ rel2,
    const int* __restrict__ mask, float* __restrict__ out) {
    // per-wave P tile, 96-byte row stride (16B aligned, limited bank conflicts)
    __shared__ _Float16 pshare[8][16][48];

    const int wslot = threadIdx.x >> 5;
    const int lane  = threadIdx.x & 31;
    const int wid   = blockIdx.x * 8 + wslot;     // 3072 wave tasks
    const int qtile = wid & 63;                   // S/16
    const int bh    = wid >> 6;                   // b*NH + h
    const int b     = bh / NHH;
    const int h     = bh % NHH;
    const int g  = lane >> 4;
    const int nl = lane & 15;
    const int q0 = qtile * 16;

    const _Float16* Qb  = q16 + ((size_t)bh * SS + q0) * HDD;
    const _Float16* Kb  = k16 + (size_t)bh * SS * HDD;
    const _Float16* Vtb = vt16 + (size_t)bh * HDD * SS;
    const float* relb  = rel  + (size_t)bh * SS * SS + (size_t)q0 * SS;
    const float* rel2b = rel2 + (size_t)bh * SS * SS + (size_t)q0 * SS;
    const int* mb = mask + b * SS;

    // preload Q A-frags for the two 32-wide head-dim slabs
    v16h qa[2];
    {
        const _Float16* qrow = Qb + (size_t)nl * HDD;
#pragma unroll
        for (int s = 0; s < 2; s++)
            qa[s] = load_a_frag(qrow + s * 32, g);
    }

    v8f o[4] = {};
    float mrow[8], lrow[8];
#pragma unroll
    for (int r = 0; r < 8; r++) { mrow[r] = -3.0e38f; lrow[r] = 0.0f; }

    const float sc = 0.125f;             // 1/sqrt(64)
    const float L2E = 1.44269504f;

    // prologue: K frags for the first 32-key chunk (kf[kt*2+s])
    v16h kf[4];
#pragma unroll
    for (int kt = 0; kt < 2; kt++)
#pragma unroll
        for (int s = 0; s < 2; s++)
            kf[kt * 2 + s] = *(const v16h*)(Kb + (size_t)(kt * 16 + nl) * HDD +
                                            s * 32 + g * 16);

    for (int k0 = 0; k0 < SS; k0 += 32) {
        // ---- issue this iteration's latency-critical loads first ----
        v16h vf[4];                         // V^T frags for the P@V at the end
#pragma unroll
        for (int ht = 0; ht < 4; ht++)
            vf[ht] = *(const v16h*)(Vtb + (size_t)(ht * 16 + nl) * SS + k0 + g * 16);

        float r1v[2][8], r2v[2][8];         // rel/rel2 tiles (the HBM-bound data)
        int mk[2];
#pragma unroll
        for (int kt = 0; kt < 2; kt++) {
            const int key = k0 + kt * 16 + nl;
            mk[kt] = mb[key];
#pragma unroll
            for (int r = 0; r < 8; r++) {
                const int qr = 8 * g + r;
                r1v[kt][r] = relb[(size_t)qr * SS + key];
                r2v[kt][r] = rel2b[(size_t)qr * SS + key];
            }
        }

        // ---- scores with already-resident K frags ----
        v8f st[2] = {};
#pragma unroll
        for (int kt = 0; kt < 2; kt++)
#pragma unroll
            for (int s = 0; s < 2; s++)
                st[kt] = wmma16(qa[s], kf[kt * 2 + s], st[kt]);

        // ---- preload next iteration's K frags; prefetch next rel lines ----
        v16h kfn[4] = {};
        if (k0 + 32 < SS) {
#pragma unroll
            for (int kt = 0; kt < 2; kt++)
#pragma unroll
                for (int s = 0; s < 2; s++)
                    kfn[kt * 2 + s] = *(const v16h*)(
                        Kb + (size_t)(k0 + 32 + kt * 16 + nl) * HDD + s * 32 + g * 16);
            const float* pf = g ? rel2b : relb;
            __builtin_prefetch(pf + (size_t)nl * SS + k0 + 32, 0, 1);
        }

        // ---- scale + rel biases + mask (C layout: row = 8g+r, col = key) ----
        float sv[2][8];
#pragma unroll
        for (int kt = 0; kt < 2; kt++) {
            const bool msk = mk[kt] != 0;
#pragma unroll
            for (int r = 0; r < 8; r++) {
                float v = st[kt][r] * sc + r1v[kt][r] + r2v[kt][r];
                sv[kt][r] = msk ? -1.0e10f : v;
            }
        }

        // ---- online softmax: reduce over the 16 lanes holding each row ----
        float alpha[8];
#pragma unroll
        for (int r = 0; r < 8; r++) {
            float mx = fmaxf(sv[0][r], sv[1][r]);
#pragma unroll
            for (int off = 1; off < 16; off <<= 1)
                mx = fmaxf(mx, __shfl_xor(mx, off, 32));
            const float mnew = fmaxf(mrow[r], mx);
            alpha[r] = exp2f((mrow[r] - mnew) * L2E);
            mrow[r] = mnew;
            float p0 = exp2f((sv[0][r] - mnew) * L2E);
            float p1 = exp2f((sv[1][r] - mnew) * L2E);
            sv[0][r] = p0; sv[1][r] = p1;
            float ps = p0 + p1;
#pragma unroll
            for (int off = 1; off < 16; off <<= 1)
                ps += __shfl_xor(ps, off, 32);
            lrow[r] = lrow[r] * alpha[r] + ps;
        }

        // rescale accumulator
#pragma unroll
        for (int ht = 0; ht < 4; ht++)
#pragma unroll
            for (int r = 0; r < 8; r++)
                o[ht][r] *= alpha[r];

        // ---- P: C layout -> A layout via LDS (same-wave bounce) ----
#pragma unroll
        for (int r = 0; r < 8; r++) {
            pshare[wslot][8 * g + r][nl]      = (_Float16)sv[0][r];
            pshare[wslot][8 * g + r][16 + nl] = (_Float16)sv[1][r];
        }
        v16h pa = load_a_frag(&pshare[wslot][nl][0], g);

        // ---- P(16x32) @ V(32x64): 4 WMMAs, V^T rows are contiguous ----
#pragma unroll
        for (int ht = 0; ht < 4; ht++)
            o[ht] = wmma16(pa, vf[ht], o[ht]);

        // rotate K frags
#pragma unroll
        for (int i = 0; i < 4; i++) kf[i] = kfn[i];
    }

    // ---- normalize and store merged-head fp32 output ----
    float invl[8];
#pragma unroll
    for (int r = 0; r < 8; r++) invl[r] = 1.0f / lrow[r];
    float* ob = out + ((size_t)b * SS + q0) * HH + h * HDD;
#pragma unroll
    for (int ht = 0; ht < 4; ht++)
#pragma unroll
        for (int r = 0; r < 8; r++)
            ob[(size_t)(8 * g + r) * HH + ht * 16 + nl] = o[ht][r] * invl[r];
}

// ---------------- launch ----------------
extern "C" void kernel_launch(void* const* d_in, const int* in_sizes, int n_in,
                              void* d_out, int out_size, void* d_ws, size_t ws_size,
                              hipStream_t stream) {
    const float* hs   = (const float*)d_in[0];
    const float* Wq   = (const float*)d_in[1];
    const float* bq   = (const float*)d_in[2];
    const float* Wk   = (const float*)d_in[3];
    const float* bk   = (const float*)d_in[4];
    const float* Wv   = (const float*)d_in[5];
    const float* bv   = (const float*)d_in[6];
    const float* rel  = (const float*)d_in[7];
    const float* rel2 = (const float*)d_in[8];
    const int*   msk  = (const int*)d_in[9];
    float* out = (float*)d_out;

    // workspace layout (f16), offsets multiple of 256B; total ~28.7 MB
    char* ws = (char*)d_ws;
    const size_t HS_B = (size_t)BB * SS * HH * 2;        // 6,291,456
    const size_t WT_B = (size_t)3 * HH * HH * 2;         // 3,538,944
    const size_t QK_B = (size_t)BB * NHH * SS * HDD * 2; // 6,291,456
    _Float16* hs16 = (_Float16*)(ws);
    _Float16* wt16 = (_Float16*)(ws + HS_B);
    _Float16* q16  = (_Float16*)(ws + HS_B + WT_B);
    _Float16* k16  = (_Float16*)(ws + HS_B + WT_B + QK_B);
    _Float16* vt16 = (_Float16*)(ws + HS_B + WT_B + 2 * QK_B);

    prep_hs<<<(BB * SS * HH) / 256, 256, 0, stream>>>(hs, hs16);
    prep_w<<<(3 * HH * HH) / 256, 256, 0, stream>>>(Wq, Wk, Wv, wt16);
    // 3 mats * 256 m-tiles * 12 heads = 9216 waves / 8 waves per block
    qkv_gemm<<<9216 / 8, 256, 0, stream>>>(hs16, wt16, bq, bk, bv, q16, k16, vt16);
    // 4*12*64 = 3072 waves / 8 per block
    attn<<<3072 / 8, 256, 0, stream>>>(q16, k16, vt16, rel, rel2, msk, out);
}
